// AttentionBlock_28020366639143
// MI455X (gfx1250) — matrix operand order repaired and checked by reading, not compiled, not run on hardware
//
#include <hip/hip_runtime.h>
#include <hip/hip_bf16.h>
#include <cstdint>

// ---------------------------------------------------------------------------
// AttentionBlock for MI455X (gfx1250, wave32, WMMA).
// B=4, C=H=W=S=256.
//
// Pipeline:
//   k_cvt3 : Wq/Wk/Wv fp32 -> f16 (workspace)
//   k_proj : q/k/v = conv1x1 (GEMM, v_wmma_f32_16x16x32_f16), stored f16 in ws
//   k_attn : per (b,i): scores = Q@K, softmax, out = sigmoid(A@V / rowsum)
//            Q/K^T/V^T staged in LDS (~272KB of the 320KB WGP LDS),
//            Q staged with global_load_async_to_lds_b128 (ASYNCcnt) that
//            overlaps the K^T transpose staging.
//
// Workspace layout (needs ~385MB):
//   [0          , 393216)           : W16 q|k|v   (3 * 256*256 f16)
//   [524288     , +134217728)       : q16  [4][256][65536] f16
//   [+134217728 , +268435456)       : k16
//   [+268435456 , +402653184)       : v16
// ---------------------------------------------------------------------------

#define CC   256
#define SS   256
#define HW   (SS*SS)          // 65536
#define PADK 264              // padded f16 row stride (x2B = 528, 16B multiple)

typedef __attribute__((ext_vector_type(16))) _Float16 v16h;
typedef __attribute__((ext_vector_type(8)))  _Float16 v8h;
typedef __attribute__((ext_vector_type(4)))  _Float16 v4h;
typedef __attribute__((ext_vector_type(8)))  float    v8f;

// Build a 16xf16 WMMA operand from two 8xf16 chunks (VGPRs 0-3 = lo, 4-7 = hi).
__device__ __forceinline__ v16h make_frag(const _Float16* p_lo, const _Float16* p_hi) {
    union { v16h v; v8h h[2]; } u;
    u.h[0] = *(const v8h*)p_lo;
    u.h[1] = *(const v8h*)p_hi;
    return u.v;
}

// A operand, row-major [m][k] with row stride `stride` (f16 units).
// Lane L: m = L&15, half = L>>4 ; chunks at k0+8*half and k0+16+8*half.
__device__ __forceinline__ v16h frag_a(const _Float16* base, int stride, int m,
                                       int k0, int lane) {
    const _Float16* p = base + (size_t)m * stride + k0 + 8 * (lane >> 4);
    return make_frag(p, p + 16);
}

// B operand, staged TRANSPOSED as [n][k] with row stride `stride`.
// Lane L: n = L&15, half = L>>4 ; chunks at k0+8*half and k0+16+8*half.
__device__ __forceinline__ v16h frag_bt(const _Float16* base, int stride, int n,
                                        int k0, int lane) {
    const _Float16* p = base + (size_t)n * stride + k0 + 8 * (lane >> 4);
    return make_frag(p, p + 16);
}

__device__ __forceinline__ v8f wmma_f16(v16h a, v16h b, v8f c) {
    return __builtin_amdgcn_wmma_f32_16x16x32_f16(false, a, false, b,
                                                  (short)0, c, false, false);
}

// ---------------------------------------------------------------------------
// Kernel 0: convert the three 256x256 fp32 weight matrices to f16.
// ---------------------------------------------------------------------------
__global__ void k_cvt3(const float* __restrict__ wq, const float* __restrict__ wk,
                       const float* __restrict__ wv, _Float16* __restrict__ o) {
    int idx = blockIdx.x * 256 + threadIdx.x;
    if (idx >= 3 * CC * CC) return;
    int which = idx / (CC * CC), r = idx % (CC * CC);
    const float* s = (which == 0) ? wq : (which == 1) ? wk : wv;
    o[idx] = (_Float16)s[r];
}

// ---------------------------------------------------------------------------
// Kernel 1: y16[b][o][hw] = f16( W16[o][:] . x[b][:][hw] + bias[o] )
// Grid: (HW/128, B). Block: 256 (8 waves). Per-WG tile: 256(o) x 128(hw).
// Wave grid 4(M) x 2(N): 64(o) x 64(hw) per wave -> 4x4 accum tiles.
// BT (x-tile transposed [hw][c], 67.6KB) in dynamic LDS.
// ---------------------------------------------------------------------------
__global__ __launch_bounds__(256) void
k_proj(const float* __restrict__ x, const _Float16* __restrict__ W16,
       const float* __restrict__ bias, _Float16* __restrict__ y16) {
    extern __shared__ _Float16 BT[];     // [128][PADK]

    const int b   = blockIdx.y;
    const int hw0 = blockIdx.x * 128;
    const int tid = threadIdx.x;

    // Stage B^T: read 4 channel rows per thread (coalesced in hw), pack 4 f16,
    // b64 scatter store into BT[hw][c0..c0+3].
    for (int it = 0; it < 32; ++it) {
        int idx = it * 256 + tid;        // 8192 packed items
        int hw  = idx & 127;
        int c0  = (idx >> 7) * 4;        // 0,4,...,252
        const float* src = x + ((size_t)b * CC + c0) * HW + hw0 + hw;
        v4h p;
#pragma unroll
        for (int u = 0; u < 4; ++u) p[u] = (_Float16)src[(size_t)u * HW];
        *(v4h*)&BT[hw * PADK + c0] = p;
    }
    __syncthreads();

    const int wave = tid >> 5, lane = tid & 31;
    const int wm = wave >> 1, wn = wave & 1;        // 4 x 2 wave grid
    const int m_base = wm * 64, n_base = wn * 64;
    const int lm = lane & 15, lh = lane >> 4;

    v8f acc[4][4];
#pragma unroll
    for (int mt = 0; mt < 4; ++mt)
#pragma unroll
        for (int nt = 0; nt < 4; ++nt) acc[mt][nt] = (v8f)0.0f;

    for (int k0 = 0; k0 < CC; k0 += 32) {
        v16h af[4];
#pragma unroll
        for (int mt = 0; mt < 4; ++mt)
            af[mt] = frag_a(W16, CC, m_base + mt * 16 + lm, k0, lane);
#pragma unroll
        for (int nt = 0; nt < 4; ++nt) {
            v16h bf = frag_bt(BT, PADK, n_base + nt * 16 + lm, k0, lane);
#pragma unroll
            for (int mt = 0; mt < 4; ++mt)
                acc[mt][nt] = wmma_f16(af[mt], bf, acc[mt][nt]);
        }
    }

    // Epilogue: + bias, convert, store f16.  C/D: row = g + 8*half, col = lane&15.
#pragma unroll
    for (int mt = 0; mt < 4; ++mt) {
#pragma unroll
        for (int g = 0; g < 8; ++g) {
            int row = m_base + mt * 16 + g + 8 * lh;
            float bv = bias[row];
#pragma unroll
            for (int nt = 0; nt < 4; ++nt) {
                int col = hw0 + n_base + nt * 16 + lm;
                y16[((size_t)b * CC + row) * HW + col] =
                    (_Float16)(acc[mt][nt][g] + bv);
            }
        }
    }
}

// ---------------------------------------------------------------------------
// Kernel 2: fused attention per (b,i) slice.
// Grid: (256, 4) = (i, b). Block: 512 (16 waves, 4x4 wave grid, 64x64/wave).
// LDS: Q/P [256][264] f16 + K^T/V^T [256][264] f16 + 2 reduce buffers.
// ---------------------------------------------------------------------------
struct P2Smem {
    _Float16 sh[256 * PADK];   // Q slice, later reused for P = exp(S - max)
    _Float16 kt[256 * PADK];   // K^T slice, later reused for V^T
    float    redmax[256 * 4];
    float    redsum[256 * 4];
};

__global__ __launch_bounds__(512, 1) void
k_attn(const _Float16* __restrict__ q16, const _Float16* __restrict__ k16,
       const _Float16* __restrict__ v16, float* __restrict__ out) {
    extern __shared__ char smem_raw[];
    P2Smem& sm = *(P2Smem*)smem_raw;

    const int i = blockIdx.x, b = blockIdx.y;
    const int tid = threadIdx.x;
    const int wave = tid >> 5, lane = tid & 31;
    const int wm = wave >> 2, wn = wave & 3;        // 4 x 4 wave grid
    const int lm = lane & 15, lh = lane >> 4;

    // ---- Launch async copy Q[h][j] (row-major, contiguous) into LDS.  ----
    // sm.sh is at dynamic-LDS offset 0; each row = 512B = 32 lanes x 16B.
    // The ASYNC engine runs this while we do the K^T transpose below.
    {
        const _Float16* qbase = q16 + ((size_t)b * CC + i) * HW;
        for (int r = wave; r < 256; r += 16) {
            unsigned laddr = (unsigned)((r * PADK + lane * 8) * 2);
            const _Float16* g = qbase + r * SS + lane * 8;
            asm volatile("global_load_async_to_lds_b128 %0, %1, off"
                         :: "v"(laddr), "v"(g) : "memory");
        }
    }

    // ---- Stage K^T[w][j] from k[b][j][i][w] (rows contiguous in w).   ----
    // Read 4 j-rows x 8 w per thread, pack 4 f16 along j, b64 scatter store.
    {
        const int w8 = (lane) * 8;
        const int jg = wave * 4;                 // 16 waves x 4 rows = 64/pass
        const _Float16* kbase = k16 + (size_t)b * CC * HW + (size_t)i * SS;
        for (int j0 = 0; j0 < 256; j0 += 64) {
            int j = j0 + jg;
            const _Float16* src = kbase + (size_t)j * HW + w8;
            v8h t0 = *(const v8h*)(src);
            v8h t1 = *(const v8h*)(src + HW);
            v8h t2 = *(const v8h*)(src + 2 * (size_t)HW);
            v8h t3 = *(const v8h*)(src + 3 * (size_t)HW);
#pragma unroll
            for (int e = 0; e < 8; ++e) {
                v4h p; p[0] = t0[e]; p[1] = t1[e]; p[2] = t2[e]; p[3] = t3[e];
                *(v4h*)&sm.kt[(w8 + e) * PADK + j] = p;
            }
        }
    }
    asm volatile("s_wait_asynccnt 0" ::: "memory");   // Q copy done (this wave)
    __syncthreads();

    // ---- GEMM1: S = Q (256x256) @ K (256x256), 64x64 tile per wave. ----
    v8f acc[4][4];
#pragma unroll
    for (int mt = 0; mt < 4; ++mt)
#pragma unroll
        for (int nt = 0; nt < 4; ++nt) acc[mt][nt] = (v8f)0.0f;

    for (int k0 = 0; k0 < CC; k0 += 32) {
        v16h af[4];
#pragma unroll
        for (int mt = 0; mt < 4; ++mt)
            af[mt] = frag_a(sm.sh, PADK, wm * 64 + mt * 16 + lm, k0, lane);
#pragma unroll
        for (int nt = 0; nt < 4; ++nt) {
            v16h bf = frag_bt(sm.kt, PADK, wn * 64 + nt * 16 + lm, k0, lane);
#pragma unroll
            for (int mt = 0; mt < 4; ++mt)
                acc[mt][nt] = wmma_f16(af[mt], bf, acc[mt][nt]);
        }
    }

    // ---- Softmax (unnormalized): row max -> exp -> row sum.            ----
    // C/D layout: element (g, lane) = row g+8*lh of tile, col lm.
    // Reduce over lanes 0-15 / 16-31 independently (xor<=8), combine the 4
    // N-waves through LDS.
#pragma unroll
    for (int mt = 0; mt < 4; ++mt) {
#pragma unroll
        for (int g = 0; g < 8; ++g) {
            float m = acc[mt][0][g];
#pragma unroll
            for (int nt = 1; nt < 4; ++nt) m = fmaxf(m, acc[mt][nt][g]);
#pragma unroll
            for (int d = 8; d >= 1; d >>= 1) m = fmaxf(m, __shfl_xor(m, d, 32));
            if (lm == 0) {
                int r = wm * 64 + mt * 16 + g + 8 * lh;
                sm.redmax[r * 4 + wn] = m;
            }
        }
    }
    __syncthreads();

#pragma unroll
    for (int mt = 0; mt < 4; ++mt) {
#pragma unroll
        for (int g = 0; g < 8; ++g) {
            int r = wm * 64 + mt * 16 + g + 8 * lh;
            float gm = fmaxf(fmaxf(sm.redmax[r * 4 + 0], sm.redmax[r * 4 + 1]),
                             fmaxf(sm.redmax[r * 4 + 2], sm.redmax[r * 4 + 3]));
            float s = 0.0f;
#pragma unroll
            for (int nt = 0; nt < 4; ++nt) {
                float e = __expf(acc[mt][nt][g] - gm);
                acc[mt][nt][g] = e;
                s += e;
            }
#pragma unroll
            for (int d = 8; d >= 1; d >>= 1) s += __shfl_xor(s, d, 32);
            if (lm == 0) sm.redsum[r * 4 + wn] = s;
        }
    }
    __syncthreads();   // all waves done with GEMM1 + Q; sh/kt reusable now

    // ---- Write P = exp(S-max) as f16 into sh; restage V^T over kt. ----
#pragma unroll
    for (int mt = 0; mt < 4; ++mt)
#pragma unroll
        for (int nt = 0; nt < 4; ++nt)
#pragma unroll
            for (int g = 0; g < 8; ++g) {
                int r = wm * 64 + mt * 16 + g + 8 * lh;
                sm.sh[r * PADK + wn * 64 + nt * 16 + lm] =
                    (_Float16)acc[mt][nt][g];
            }
    {
        const int w8 = (lane) * 8;
        const int jg = wave * 4;
        const _Float16* vbase = v16 + ((size_t)b * CC + i) * HW;
        for (int j0 = 0; j0 < 256; j0 += 64) {
            int j = j0 + jg;
            const _Float16* src = vbase + (size_t)j * SS + w8;   // v[b][i][j][w]
            v8h t0 = *(const v8h*)(src);
            v8h t1 = *(const v8h*)(src + SS);
            v8h t2 = *(const v8h*)(src + 2 * SS);
            v8h t3 = *(const v8h*)(src + 3 * SS);
#pragma unroll
            for (int e = 0; e < 8; ++e) {
                v4h p; p[0] = t0[e]; p[1] = t1[e]; p[2] = t2[e]; p[3] = t3[e];
                *(v4h*)&sm.kt[(w8 + e) * PADK + j] = p;
            }
        }
    }
    __syncthreads();

    // ---- GEMM2: O = P (256x256) @ V (256x256). ----
#pragma unroll
    for (int mt = 0; mt < 4; ++mt)
#pragma unroll
        for (int nt = 0; nt < 4; ++nt) acc[mt][nt] = (v8f)0.0f;

    for (int k0 = 0; k0 < CC; k0 += 32) {
        v16h af[4];
#pragma unroll
        for (int mt = 0; mt < 4; ++mt)
            af[mt] = frag_a(sm.sh, PADK, wm * 64 + mt * 16 + lm, k0, lane);
#pragma unroll
        for (int nt = 0; nt < 4; ++nt) {
            v16h bf = frag_bt(sm.kt, PADK, wn * 64 + nt * 16 + lm, k0, lane);
#pragma unroll
            for (int mt = 0; mt < 4; ++mt)
                acc[mt][nt] = wmma_f16(af[mt], bf, acc[mt][nt]);
        }
    }

    // ---- Epilogue: divide by row sum, sigmoid, store fp32. ----
#pragma unroll
    for (int mt = 0; mt < 4; ++mt) {
#pragma unroll
        for (int g = 0; g < 8; ++g) {
            int r = wm * 64 + mt * 16 + g + 8 * lh;
            float rs = sm.redsum[r * 4 + 0] + sm.redsum[r * 4 + 1] +
                       sm.redsum[r * 4 + 2] + sm.redsum[r * 4 + 3];
            float inv = 1.0f / rs;
#pragma unroll
            for (int nt = 0; nt < 4; ++nt) {
                int col = wn * 64 + nt * 16 + lm;
                float v = acc[mt][nt][g] * inv;
                out[(((size_t)b * CC + i) * SS + r) * SS + col] =
                    1.0f / (1.0f + __expf(-v));
            }
        }
    }
}

// ---------------------------------------------------------------------------
extern "C" void kernel_launch(void* const* d_in, const int* in_sizes, int n_in,
                              void* d_out, int out_size, void* d_ws, size_t ws_size,
                              hipStream_t stream) {
    (void)in_sizes; (void)n_in; (void)out_size; (void)ws_size;
    const float* query  = (const float*)d_in[0];
    const float* key_in = (const float*)d_in[1];
    const float* value  = (const float*)d_in[2];
    const float* Wq = (const float*)d_in[3];
    const float* bq = (const float*)d_in[4];
    const float* Wk = (const float*)d_in[5];
    const float* bk = (const float*)d_in[6];
    const float* Wv = (const float*)d_in[7];
    const float* bv = (const float*)d_in[8];
    float* out = (float*)d_out;

    char* ws = (char*)d_ws;
    _Float16* W16  = (_Float16*)ws;                      // 3 * 128KB
    _Float16* W16q = W16;
    _Float16* W16k = W16 + CC * CC;
    _Float16* W16v = W16 + 2 * CC * CC;
    size_t qkv_elems = (size_t)4 * CC * HW;              // 64M f16 = 128MB
    _Float16* q16 = (_Float16*)(ws + 524288);
    _Float16* k16 = q16 + qkv_elems;
    _Float16* v16 = k16 + qkv_elems;

    // 1) weights -> f16
    k_cvt3<<<dim3((3 * CC * CC + 255) / 256), dim3(256), 0, stream>>>(Wq, Wk, Wv, W16);

    // 2) QKV projections (GEMM + bias), f16 outputs; BT in dynamic LDS
    dim3 pg(HW / 128, 4);
    size_t proj_lds = (size_t)128 * PADK * sizeof(_Float16);
    k_proj<<<pg, dim3(256), proj_lds, stream>>>(query,  W16q, bq, q16);
    k_proj<<<pg, dim3(256), proj_lds, stream>>>(key_in, W16k, bk, k16);
    k_proj<<<pg, dim3(256), proj_lds, stream>>>(value,  W16v, bv, v16);

    // 3) fused attention: one workgroup per (b,i); ~272KB dynamic LDS
    k_attn<<<dim3(256, 4), dim3(512), sizeof(P2Smem), stream>>>(q16, k16, v16, out);
}
